// DMPNNLayer_23295902613716
// MI455X (gfx1250) — matrix operand order repaired and checked by reading, not compiled, not run on hardware
//
#include <hip/hip_runtime.h>

#define NN 50000
#define NE 800000
#define OUTF 64

typedef __bf16 v16bf __attribute__((ext_vector_type(16)));
typedef float  v8f   __attribute__((ext_vector_type(8)));

union ABf { unsigned int u[8]; v16bf v; };

__device__ inline unsigned int pack2(float x, float y) {
    union { __bf16 h[2]; unsigned int u; } p;
    p.h[0] = (__bf16)x; p.h[1] = (__bf16)y;
    return p.u;
}

// ---------------------------------------------------------------- zero full
__global__ __launch_bounds__(256) void zero_full(float* __restrict__ full) {
    size_t i = ((size_t)blockIdx.x * 256 + threadIdx.x) * 4;
    float4 z; z.x = 0.f; z.y = 0.f; z.z = 0.f; z.w = 0.f;
    *(float4*)(full + i) = z;
}

// ---------------------------------------------------------------- messages (WMMA)
// Each block: 8 waves x 16 edges = 128 edges.
// direct   = [edge_feats | node_feats[src]] @ W + b   -> stored in new_backward slot, scatter-added to full[dst]
// backward = [edge_feats | node_feats[dst]] @ W + b   -> stored in new_direct slot
__global__ __launch_bounds__(256)
void dmpnn_msg(const float* __restrict__ node_feats,
               const float* __restrict__ edge_feats,
               const float* __restrict__ W,
               const float* __restrict__ bias,
               const int*   __restrict__ src,
               const int*   __restrict__ dst,
               float* __restrict__ out_nd,   // new_direct slot   <- backward
               float* __restrict__ out_nb,   // new_backward slot <- direct
               float* __restrict__ full)
{
    __shared__ unsigned int lds_W[96 * 64];  // bf16-pair packed W: row kp = (K 2kp,2kp+1), col n
    __shared__ float lds_b[64];
    __shared__ int   lds_s[128];
    __shared__ int   lds_d[128];

    const int tid    = threadIdx.x;
    const int eblock = blockIdx.x * 128;

    for (int i = tid; i < 96 * 64; i += 256) {
        int kp = i >> 6, n = i & 63;
        lds_W[i] = pack2(W[(2 * kp) * 64 + n], W[(2 * kp + 1) * 64 + n]);
    }
    if (tid < 64)  lds_b[tid] = bias[tid];
    if (tid < 128) lds_s[tid] = src[eblock + tid];
    else           lds_d[tid - 128] = dst[eblock + (tid - 128)];
    __syncthreads();

    const int wave  = tid >> 5;
    const int lane  = tid & 31;
    const int row   = lane & 15;
    const int hi    = lane >> 4;          // which half of the K stripe this lane holds
    const int le    = wave * 16 + row;    // local edge index of this lane's A-row
    const int ebase = eblock + wave * 16;
    const int sidx  = lds_s[le];
    const int didx  = lds_d[le];

    const v8f vzero = {0.f, 0.f, 0.f, 0.f, 0.f, 0.f, 0.f, 0.f};
    v8f accD[4], accB[4];
#pragma unroll
    for (int nt = 0; nt < 4; ++nt) { accD[nt] = vzero; accB[nt] = vzero; }

#pragma unroll
    for (int kc = 0; kc < 6; ++kc) {      // K = 192 in chunks of 32
        ABf aS, aD;
        if (kc < 2) {                     // edge-feature part (shared by direct & backward)
            const float* p = edge_feats + (size_t)(ebase + row) * 64 + kc * 32 + hi * 8;
            float4 f0 = *(const float4*)(p);
            float4 f1 = *(const float4*)(p + 4);
            float4 f2 = *(const float4*)(p + 16);
            float4 f3 = *(const float4*)(p + 20);
            aS.u[0] = pack2(f0.x, f0.y); aS.u[1] = pack2(f0.z, f0.w);
            aS.u[2] = pack2(f1.x, f1.y); aS.u[3] = pack2(f1.z, f1.w);
            aS.u[4] = pack2(f2.x, f2.y); aS.u[5] = pack2(f2.z, f2.w);
            aS.u[6] = pack2(f3.x, f3.y); aS.u[7] = pack2(f3.z, f3.w);
            aD = aS;
        } else {                          // node-feature part (gathered by src / dst)
            const int ko = kc * 32 - 64 + hi * 8;
            const float* p = node_feats + (size_t)sidx * 128 + ko;
            float4 f0 = *(const float4*)(p);
            float4 f1 = *(const float4*)(p + 4);
            float4 f2 = *(const float4*)(p + 16);
            float4 f3 = *(const float4*)(p + 20);
            aS.u[0] = pack2(f0.x, f0.y); aS.u[1] = pack2(f0.z, f0.w);
            aS.u[2] = pack2(f1.x, f1.y); aS.u[3] = pack2(f1.z, f1.w);
            aS.u[4] = pack2(f2.x, f2.y); aS.u[5] = pack2(f2.z, f2.w);
            aS.u[6] = pack2(f3.x, f3.y); aS.u[7] = pack2(f3.z, f3.w);
            const float* q = node_feats + (size_t)didx * 128 + ko;
            float4 g0 = *(const float4*)(q);
            float4 g1 = *(const float4*)(q + 4);
            float4 g2 = *(const float4*)(q + 16);
            float4 g3 = *(const float4*)(q + 20);
            aD.u[0] = pack2(g0.x, g0.y); aD.u[1] = pack2(g0.z, g0.w);
            aD.u[2] = pack2(g1.x, g1.y); aD.u[3] = pack2(g1.z, g1.w);
            aD.u[4] = pack2(g2.x, g2.y); aD.u[5] = pack2(g2.z, g2.w);
            aD.u[6] = pack2(g3.x, g3.y); aD.u[7] = pack2(g3.z, g3.w);
        }

        const int wrow0 = kc * 16 + hi * 4;  // packed-pair row base in lds_W
        const int ncb   = lane & 15;
#pragma unroll
        for (int nt = 0; nt < 4; ++nt) {
            ABf bf;
            const int ncol = nt * 16 + ncb;
#pragma unroll
            for (int j = 0; j < 8; ++j) bf.u[j] = lds_W[(wrow0 + j + (j >= 4 ? 4 : 0)) * 64 + ncol];
            accD[nt] = __builtin_amdgcn_wmma_f32_16x16x32_bf16(
                false, aS.v, false, bf.v, (short)0, accD[nt], false, false);
            accB[nt] = __builtin_amdgcn_wmma_f32_16x16x32_bf16(
                false, aD.v, false, bf.v, (short)0, accB[nt], false, false);
        }
    }

    // Epilogue: bias, store messages, scatter-add direct into full[dst]
#pragma unroll
    for (int nt = 0; nt < 4; ++nt) {
        const int n  = nt * 16 + (lane & 15);
        const float bb = lds_b[n];
#pragma unroll
        for (int r = 0; r < 8; ++r) {
            const int m  = r + hi * 8;          // C layout: lane<16 -> M=r, lane>=16 -> M=8+r
            const int eg = ebase + m;
            const float dval = accD[nt][r] + bb;   // direct message
            const float bval = accB[nt][r] + bb;   // backward message
            out_nb[(size_t)eg * 64 + n] = dval;
            out_nd[(size_t)eg * 64 + n] = bval;
            atomicAdd(&full[(size_t)lds_d[wave * 16 + m] * 64 + n], dval);
        }
    }
}

// ---------------------------------------------------------------- relu
__global__ __launch_bounds__(256)
void relu_k(const float* __restrict__ full, float* __restrict__ out) {
    size_t i = ((size_t)blockIdx.x * 256 + threadIdx.x) * 4;
    float4 v = *(const float4*)(full + i);
    v.x = v.x > 0.f ? v.x : 0.f;
    v.y = v.y > 0.f ? v.y : 0.f;
    v.z = v.z > 0.f ? v.z : 0.f;
    v.w = v.w > 0.f ? v.w : 0.f;
    *(float4*)(out + i) = v;
}

// ---------------------------------------------------------------- edge update
// new_direct   = full[src] - backward   (backward currently stored in out_nd)
// new_backward = full[dst] - direct     (direct   currently stored in out_nb)
__global__ __launch_bounds__(256)
void edge_update(const int* __restrict__ src, const int* __restrict__ dst,
                 const float* __restrict__ full,
                 float* __restrict__ out_nd, float* __restrict__ out_nb) {
    size_t gid = (size_t)blockIdx.x * 256 + threadIdx.x;
    int e  = (int)(gid >> 4);
    int n4 = (int)((gid & 15) << 2);
    int s = src[e], d = dst[e];
    float4 fs = *(const float4*)(full + (size_t)s * 64 + n4);
    float4 fd = *(const float4*)(full + (size_t)d * 64 + n4);
    size_t off = (size_t)e * 64 + n4;
    float4 bk  = *(const float4*)(out_nd + off);
    float4 dir = *(const float4*)(out_nb + off);
    float4 r0, r1;
    r0.x = fs.x - bk.x;  r0.y = fs.y - bk.y;  r0.z = fs.z - bk.z;  r0.w = fs.w - bk.w;
    r1.x = fd.x - dir.x; r1.y = fd.y - dir.y; r1.z = fd.z - dir.z; r1.w = fd.w - dir.w;
    *(float4*)(out_nd + off) = r0;
    *(float4*)(out_nb + off) = r1;
}

// ---------------------------------------------------------------- launch
extern "C" void kernel_launch(void* const* d_in, const int* in_sizes, int n_in,
                              void* d_out, int out_size, void* d_ws, size_t ws_size,
                              hipStream_t stream) {
    const float* node_feats = (const float*)d_in[0];
    const float* edge_feats = (const float*)d_in[1];
    const float* W          = (const float*)d_in[2];
    const float* b          = (const float*)d_in[3];
    const int*   src        = (const int*)d_in[4];
    const int*   dst        = (const int*)d_in[5];

    float* out      = (float*)d_out;
    float* out_node = out;                                     // [NN, 64]
    float* out_nd   = out + (size_t)NN * OUTF;                 // [NE, 64]
    float* out_nb   = out_nd + (size_t)NE * OUTF;              // [NE, 64]
    float* full     = (float*)d_ws;                            // [NN, 64] scratch

    // 50000*64/4 = 800000 float4 slots -> 3125 blocks
    zero_full<<<3125, 256, 0, stream>>>(full);
    // 800000 edges / 128 per block
    dmpnn_msg<<<6250, 256, 0, stream>>>(node_feats, edge_feats, W, b, src, dst,
                                        out_nd, out_nb, full);
    relu_k<<<3125, 256, 0, stream>>>(full, out_node);
    // 800000 edges * 16 float4-chunks / 256 = 50000 blocks
    edge_update<<<50000, 256, 0, stream>>>(src, dst, full, out_nd, out_nb);
}
